// OCA_5549097747036
// MI455X (gfx1250) — matrix operand order repaired and checked by reading, not compiled, not run on hardware
//
#include <hip/hip_runtime.h>
#include <stdint.h>

typedef __bf16 bf16;
typedef __attribute__((ext_vector_type(8)))  bf16  bf16x8;
typedef __attribute__((ext_vector_type(16))) bf16  bf16x16;
typedef __attribute__((ext_vector_type(8)))  float f32x8;

#define IMG   256
#define NPIX  65536
#define SCALEQ 0.25f   // 16^-0.5

static __device__ __forceinline__ f32x8 wmma_bf16(bf16x16 a, bf16x16 b, f32x8 c) {
  return __builtin_amdgcn_wmma_f32_16x16x32_bf16(false, a, false, b, (short)0, c, false, false);
}

// ---------------------------------------------------------------------------
// Kernel 1: QKV 1x1 conv.  out = W(288x96) @ x(96 x N) + b
//   q -> qs (pixel-major bf16, pre-scaled), k -> ks (pixel-major bf16),
//   v -> vs (channel-major bf16, feeds async halo staging)
// grid = 1024 (2 batches x 512 pixel tiles of 128), block = 256 (8 waves x 16 px)
// ---------------------------------------------------------------------------
__global__ __launch_bounds__(256) void k_qkv(
    const float* __restrict__ x, const float* __restrict__ w,
    const float* __restrict__ bias,
    bf16* __restrict__ qs, bf16* __restrict__ ks, bf16* __restrict__ vs)
{
  __shared__ __align__(16) bf16 xs[128][104];  // [pixel][channel]
  __shared__ __align__(16) bf16 wl[96][104];   // one 96-row phase of W

  const int tid  = threadIdx.x;
  const int lane = tid & 31;
  const int wv   = tid >> 5;            // wave 0..7
  const int blk  = blockIdx.x;          // 0..1023
  const int bb   = blk >> 9;            // batch
  const int pbase = (blk & 511) * 128;
  const size_t xb = (size_t)bb * 96 * NPIX;

  for (int i = tid; i < 128 * 96; i += 256) {
    const int pix = i & 127, c = i >> 7;
    xs[pix][c] = (bf16)x[xb + (size_t)c * NPIX + pbase + pix];
  }

  const int m  = lane & 15;
  const int g  = lane >> 4;
  const int g8 = g * 8;
  const int pix = pbase + wv * 16 + m;
  const size_t pixrow = ((size_t)bb * NPIX + pix) * 96;

  #pragma unroll
  for (int phase = 0; phase < 3; ++phase) {
    __syncthreads();
    for (int i = tid; i < 96 * 96; i += 256) {
      const int r = i / 96, c = i - r * 96;
      wl[r][c] = (bf16)w[(phase * 96 + r) * 96 + c];
    }
    __syncthreads();

    for (int mt = 0; mt < 6; ++mt) {
      f32x8 acc = {0.f,0.f,0.f,0.f,0.f,0.f,0.f,0.f};
      #pragma unroll
      for (int kt = 0; kt < 3; ++kt) {
        bf16x16 af;
        {
          const bf16* ap = &wl[mt * 16 + m][kt * 32 + g8];
          const bf16x8 lo = *(const bf16x8*)ap;
          const bf16x8 hi = *(const bf16x8*)(ap + 16);
          #pragma unroll
          for (int j = 0; j < 8; ++j) { af[j] = lo[j]; af[j + 8] = hi[j]; }
        }
        bf16x16 bx;
        {
          const bf16* bp = &xs[wv * 16 + m][kt * 32 + g8];
          const bf16x8 lo = *(const bf16x8*)bp;
          const bf16x8 hi = *(const bf16x8*)(bp + 16);
          #pragma unroll
          for (int j = 0; j < 8; ++j) { bx[j] = lo[j]; bx[j + 8] = hi[j]; }
        }
        acc = wmma_bf16(af, bx, acc);
      }
      const float* bp = bias + phase * 96 + mt * 16 + g8;
      if (phase == 0) {           // Q: pixel-major, pre-scaled, vector store
        bf16x8 ov;
        #pragma unroll
        for (int v = 0; v < 8; ++v) ov[v] = (bf16)((acc[v] + bp[v]) * SCALEQ);
        *(bf16x8*)(qs + pixrow + mt * 16 + g8) = ov;
      } else if (phase == 1) {    // K: pixel-major vector store
        bf16x8 ov;
        #pragma unroll
        for (int v = 0; v < 8; ++v) ov[v] = (bf16)(acc[v] + bp[v]);
        *(bf16x8*)(ks + pixrow + mt * 16 + g8) = ov;
      } else {                    // V: channel-major (rows coalesce across lanes)
        #pragma unroll
        for (int v = 0; v < 8; ++v)
          vs[xb + (size_t)(mt * 16 + 8 * g + v) * NPIX + pix] = (bf16)(acc[v] + bp[v]);
      }
    }
  }
}

// ---------------------------------------------------------------------------
// Kernel 2: windowed attention with halo + relative position bias.
// grid = 2048 windows, block = 128 (4 waves). 24 tasks = 6 heads x 4 q-row-tiles.
// V halo is staged global->LDS with async-to-LDS copies (ASYNCcnt).
// ---------------------------------------------------------------------------
__global__ __launch_bounds__(128) void k_attn(
    const bf16* __restrict__ qs, const bf16* __restrict__ ks,
    const bf16* __restrict__ vs,
    const float* __restrict__ relh, const float* __restrict__ relw,
    bf16* __restrict__ ob)
{
  __shared__ __align__(16) bf16 v_lds[96][168];      // [channel][key]; 144..159 zero
  __shared__ __align__(16) bf16 p_lds[4][16][160];   // per-wave softmax probs
  __shared__ __align__(16) bf16 l23w[4][16][24];     // per-wave rel-width logits
  __shared__ __align__(16) bf16 l23h[4][16][24];     // per-wave rel-height logits
  __shared__ __align__(16) bf16 rw_l[32][16];
  __shared__ __align__(16) bf16 rh_l[32][16];

  const int tid  = threadIdx.x;
  const int lane = tid & 31;
  const int wv   = tid >> 5;              // wave 0..3
  const int blk  = blockIdx.x;
  const int bb   = blk >> 10;
  const int wy   = (blk & 1023) >> 5;
  const int wx   = blk & 31;
  const size_t cb = (size_t)bb * 96 * NPIX;

  for (int i = tid; i < 512; i += 128) {
    const int r = i >> 4, d = i & 15;
    rw_l[r][d] = (bf16)(r < 23 ? relw[r * 16 + d] : 0.f);
    rh_l[r][d] = (bf16)(r < 23 ? relh[r * 16 + d] : 0.f);
  }
  // zero K-pad keys 144..159 of V
  for (int i = tid; i < 96 * 8; i += 128) {
    const int c = i >> 3, kk = (i & 7) * 2;
    *(unsigned*)&v_lds[c][144 + kk] = 0u;
  }
  // async halo copy: vs (channel-major) -> v_lds[c][key], 2 bf16 (4B) per op.
  // gx is always even so both sides are 4B aligned; OOB pairs are zeroed via DS.
  for (int i = tid; i < 96 * 72; i += 128) {
    const int c   = i / 72;
    const int rem = i - c * 72;
    const int ky  = rem / 6;
    const int kx  = (rem - ky * 6) * 2;
    const int gy  = wy * 8 - 2 + ky;
    const int gx  = wx * 8 - 2 + kx;
    if (gy >= 0 && gy < IMG && gx >= 0 && gx < IMG) {
      const unsigned loff = (unsigned)(uintptr_t)&v_lds[c][ky * 12 + kx];
      const unsigned long long gaddr =
          (unsigned long long)(uintptr_t)(vs + cb + (size_t)c * NPIX + gy * IMG + gx);
      asm volatile("global_load_async_to_lds_b32 %0, %1, off"
                   :: "v"(loff), "v"(gaddr) : "memory");
    } else {
      *(unsigned*)&v_lds[c][ky * 12 + kx] = 0u;
    }
  }
  for (int i = tid; i < 4 * 16 * 16; i += 128) {   // zero K-pad columns of P
    p_lds[i >> 8][(i >> 4) & 15][144 + (i & 15)] = (bf16)0.f;
  }
  asm volatile("s_wait_asynccnt 0" ::: "memory");
  __syncthreads();

  const int m  = lane & 15;
  const int g  = lane >> 4;
  const int g8 = g * 8;

  for (int task = wv; task < 24; task += 4) {
    const int h  = task >> 2;
    const int mt = task & 3;

    // ---- A fragment: 16 q rows of this tile, K = d(16) zero-padded to 32 ----
    bf16x16 aq;
    {
      const int qrow = mt * 16 + m;
      const int gy = wy * 8 + (qrow >> 3), gx = wx * 8 + (qrow & 7);
      const bf16* qp = qs + ((size_t)bb * NPIX + gy * IMG + gx) * 96 + h * 16;
      const bf16x8 ql = *(const bf16x8*)qp;
      const bf16x8 qh = *(const bf16x8*)(qp + 8);
      #pragma unroll
      for (int j = 0; j < 8; ++j) { aq[j] = g == 0 ? ql[j] : (bf16)0.f; aq[j + 8] = g == 0 ? qh[j] : (bf16)0.f; }
    }

    // ---- rel-position logits L23 = Q @ Rel^T via WMMA ----
    #pragma unroll
    for (int nt = 0; nt < 2; ++nt) {
      bf16x16 bw, bh;
      if (g == 0) {
        const bf16* pw = &rw_l[nt * 16 + m][0];
        const bf16* ph = &rh_l[nt * 16 + m][0];
        const bf16x8 w0 = *(const bf16x8*)pw, w1 = *(const bf16x8*)(pw + 8);
        const bf16x8 h0 = *(const bf16x8*)ph, h1 = *(const bf16x8*)(ph + 8);
        #pragma unroll
        for (int j = 0; j < 8; ++j) {
          bw[j] = w0[j]; bw[j + 8] = w1[j];
          bh[j] = h0[j]; bh[j + 8] = h1[j];
        }
      } else {
        #pragma unroll
        for (int j = 0; j < 16; ++j) { bw[j] = (bf16)0.f; bh[j] = (bf16)0.f; }
      }
      const f32x8 z = {0.f,0.f,0.f,0.f,0.f,0.f,0.f,0.f};
      const f32x8 dw = wmma_bf16(aq, bw, z);
      const f32x8 dh = wmma_bf16(aq, bh, z);
      const int r = nt * 16 + m;
      if (r < 24) {
        #pragma unroll
        for (int v = 0; v < 8; ++v) {
          l23w[wv][v + 8 * g][r] = (bf16)dw[v];
          l23h[wv][v + 8 * g][r] = (bf16)dh[v];
        }
      }
    }
    asm volatile("s_wait_dscnt 0" ::: "memory");

    // ---- S = Q @ K^T (K rows vector-loaded from pixel-major global) ----
    f32x8 s[9];
    #pragma unroll
    for (int nt = 0; nt < 9; ++nt) {
      bf16x16 bk;
      bool have = false;
      if (g == 0) {
        const int key = nt * 16 + m;
        const int ky = key / 12, kx = key - ky * 12;
        const int gy = wy * 8 - 2 + ky, gx = wx * 8 - 2 + kx;
        if (gy >= 0 && gy < IMG && gx >= 0 && gx < IMG) {
          const bf16* kp = ks + ((size_t)bb * NPIX + gy * IMG + gx) * 96 + h * 16;
          const bf16x8 lo = *(const bf16x8*)kp;
          const bf16x8 hi = *(const bf16x8*)(kp + 8);
          #pragma unroll
          for (int j = 0; j < 8; ++j) { bk[j] = lo[j]; bk[j + 8] = hi[j]; }
          have = true;
        }
      }
      if (!have) {
        #pragma unroll
        for (int j = 0; j < 16; ++j) bk[j] = (bf16)0.f;
      }
      const f32x8 z = {0.f,0.f,0.f,0.f,0.f,0.f,0.f,0.f};
      s[nt] = wmma_bf16(aq, bk, z);
    }

    // ---- bias add + row softmax (rows live across 16-lane halves) ----
    #pragma unroll
    for (int v = 0; v < 8; ++v) {
      const int rowm = v + 8 * g;
      const int qrow = mt * 16 + rowm;
      const int qy = qrow >> 3, qx = qrow & 7;
      float mx = -1e30f;
      #pragma unroll
      for (int nt = 0; nt < 9; ++nt) {
        const int key = nt * 16 + m;
        const int ky = key / 12, kx = key - ky * 12;
        const float bv = (float)l23w[wv][rowm][kx - qx + 11]
                       + (float)l23h[wv][rowm][ky - qy + 11];
        const float sv = s[nt][v] + bv;
        s[nt][v] = sv;
        mx = fmaxf(mx, sv);
      }
      #pragma unroll
      for (int off = 1; off < 16; off <<= 1) mx = fmaxf(mx, __shfl_xor(mx, off, 32));
      float sum = 0.f;
      #pragma unroll
      for (int nt = 0; nt < 9; ++nt) {
        const float e = __expf(s[nt][v] - mx);
        s[nt][v] = e;
        sum += e;
      }
      #pragma unroll
      for (int off = 1; off < 16; off <<= 1) sum += __shfl_xor(sum, off, 32);
      const float inv = 1.f / sum;
      #pragma unroll
      for (int nt = 0; nt < 9; ++nt)
        p_lds[wv][rowm][nt * 16 + m] = (bf16)(s[nt][v] * inv);
    }
    asm volatile("s_wait_dscnt 0" ::: "memory");

    // ---- O = P @ V  (K = 144 zero-padded to 160) ----
    f32x8 o = {0.f,0.f,0.f,0.f,0.f,0.f,0.f,0.f};
    #pragma unroll
    for (int kt = 0; kt < 5; ++kt) {
      bf16x16 ap;
      {
        const bf16* pp = &p_lds[wv][m][kt * 32 + g8];
        const bf16x8 lo = *(const bf16x8*)pp;
        const bf16x8 hi = *(const bf16x8*)(pp + 16);
        #pragma unroll
        for (int j = 0; j < 8; ++j) { ap[j] = lo[j]; ap[j + 8] = hi[j]; }
      }
      bf16x16 bv;
      {
        const bf16* vp = &v_lds[h * 16 + m][kt * 32 + g8];
        const bf16x8 lo = *(const bf16x8*)vp;
        const bf16x8 hi = *(const bf16x8*)(vp + 16);
        #pragma unroll
        for (int j = 0; j < 8; ++j) { bv[j] = lo[j]; bv[j + 8] = hi[j]; }
      }
      o = wmma_bf16(ap, bv, o);
    }
    #pragma unroll
    for (int v = 0; v < 8; ++v) {
      const int qrow = mt * 16 + v + 8 * g;
      const int gy = wy * 8 + (qrow >> 3), gx = wx * 8 + (qrow & 7);
      ob[((size_t)bb * NPIX + gy * IMG + gx) * 96 + h * 16 + m] = (bf16)o[v];
    }
  }
}

// ---------------------------------------------------------------------------
// Kernel 3: output 1x1 conv. out(f32) = Wp(96x96) @ O(96 x N) + bp
// ---------------------------------------------------------------------------
__global__ __launch_bounds__(256) void k_proj(
    const bf16* __restrict__ ob, const float* __restrict__ w,
    const float* __restrict__ bias, float* __restrict__ out)
{
  __shared__ __align__(16) bf16 wl[96][104];

  const int tid  = threadIdx.x;
  const int lane = tid & 31;
  const int wv   = tid >> 5;
  const int blk  = blockIdx.x;
  const int bb   = blk >> 9;
  const int pbase = (blk & 511) * 128;

  for (int i = tid; i < 96 * 96; i += 256) {
    const int r = i / 96, c = i - r * 96;
    wl[r][c] = (bf16)w[i];
  }
  __syncthreads();

  const int m  = lane & 15;
  const int g  = lane >> 4;
  const int g8 = g * 8;
  const int pix = pbase + wv * 16 + m;
  const bf16* op = ob + ((size_t)bb * NPIX + pix) * 96;

  for (int mt = 0; mt < 6; ++mt) {
    f32x8 acc = {0.f,0.f,0.f,0.f,0.f,0.f,0.f,0.f};
    #pragma unroll
    for (int kt = 0; kt < 3; ++kt) {
      bf16x16 af;
      {
        const bf16* ap = &wl[mt * 16 + m][kt * 32 + g8];
        const bf16x8 lo = *(const bf16x8*)ap;
        const bf16x8 hi = *(const bf16x8*)(ap + 16);
        #pragma unroll
        for (int j = 0; j < 8; ++j) { af[j] = lo[j]; af[j + 8] = hi[j]; }
      }
      bf16x16 bo;
      {
        const bf16* bp = op + kt * 32 + g8;
        const bf16x8 lo = *(const bf16x8*)bp;
        const bf16x8 hi = *(const bf16x8*)(bp + 16);
        #pragma unroll
        for (int j = 0; j < 8; ++j) { bo[j] = lo[j]; bo[j + 8] = hi[j]; }
      }
      acc = wmma_bf16(af, bo, acc);
    }
    #pragma unroll
    for (int v = 0; v < 8; ++v) {
      const int oc = mt * 16 + v + 8 * g;
      out[((size_t)bb * 96 + oc) * NPIX + pix] = acc[v] + bias[oc];
    }
  }
}

// ---------------------------------------------------------------------------
extern "C" void kernel_launch(void* const* d_in, const int* in_sizes, int n_in,
                              void* d_out, int out_size, void* d_ws, size_t ws_size,
                              hipStream_t stream) {
  const float* x      = (const float*)d_in[0];
  const float* qkv_w  = (const float*)d_in[1];
  const float* qkv_b  = (const float*)d_in[2];
  const float* proj_w = (const float*)d_in[3];
  const float* proj_b = (const float*)d_in[4];
  const float* relh   = (const float*)d_in[5];
  const float* relw   = (const float*)d_in[6];

  const size_t TSZ = (size_t)2 * 96 * NPIX * sizeof(bf16);  // 25,165,824 B
  char* ws = (char*)d_ws;
  bf16* qs = (bf16*)(ws);
  bf16* ks = (bf16*)(ws + TSZ);
  bf16* vs = (bf16*)(ws + 2 * TSZ);
  bf16* ob = (bf16*)(ws + 3 * TSZ);
  // requires ws_size >= 4*TSZ = 100,663,296 bytes

  k_qkv <<<dim3(1024), dim3(256), 0, stream>>>(x, qkv_w, qkv_b, qs, ks, vs);
  k_attn<<<dim3(2048), dim3(128), 0, stream>>>(qs, ks, vs, relh, relw, ob);
  k_proj<<<dim3(1024), dim3(256), 0, stream>>>(ob, proj_w, proj_b, (float*)d_out);
}
